// MLP_69587060130194
// MI455X (gfx1250) — 8-bit matrix operand order repaired and checked by reading, compiled, not run on hardware
//
#include <hip/hip_runtime.h>
#include <hip/hip_bf16.h>
#include <stdint.h>

typedef __attribute__((ext_vector_type(8))) int v8i;

#define Q_EPS   1e-5f
#define RMS_EPS 1e-6f

#define BATCH 16384
#define DIN   784
#define KP1   832
#define DH    4096
#define DOUT  10
#define NP2   16
#define KP2   4096


__global__ __launch_bounds__(32) void k_zero2(float* p) {
    if (threadIdx.x < 2) p[threadIdx.x] = 0.f;
}

__global__ __launch_bounds__(256) void k_absum(const float* __restrict__ w, int n,
                                               float* __restrict__ out) {
    __shared__ float red[256];
    float s = 0.f;
    for (int i = blockIdx.x * 256 + threadIdx.x; i < n; i += gridDim.x * 256)
        s += fabsf(w[i]);
    red[threadIdx.x] = s; __syncthreads();
    for (int st = 128; st > 0; st >>= 1) {
        if (threadIdx.x < st) red[threadIdx.x] += red[threadIdx.x + st];
        __syncthreads();
    }
    if (threadIdx.x == 0) atomicAdd(out, red[0]);
}

__global__ __launch_bounds__(256) void k_wq1(const float* __restrict__ w1,
                                             const float* __restrict__ sum,
                                             int8_t* __restrict__ w1q) {
    int idx = blockIdx.x * 256 + threadIdx.x;
    if (idx >= DH * KP1) return;
    int r = idx / KP1, c = idx % KP1;
    float m = fmaxf(sum[0] / (float)(DH * DIN), Q_EPS);
    float q = 0.f;
    if (c < DIN) q = fminf(fmaxf(rintf(w1[r * DIN + c] / m), -1.f), 1.f);
    w1q[idx] = (int8_t)q;
}

__global__ __launch_bounds__(256) void k_wq2(const float* __restrict__ w2,
                                             const float* __restrict__ sum,
                                             int8_t* __restrict__ w2q) {
    int idx = blockIdx.x * 256 + threadIdx.x;
    if (idx >= NP2 * KP2) return;
    int r = idx / KP2, c = idx % KP2;
    float m = fmaxf(sum[0] / (float)(DOUT * DH), Q_EPS);
    float q = 0.f;
    if (r < DOUT) q = fminf(fmaxf(rintf(w2[r * KP2 + c] / m), -1.f), 1.f);
    w2q[idx] = (int8_t)q;
}

__global__ __launch_bounds__(256) void k_rowquant_x(const float* __restrict__ x,
                                                    int8_t* __restrict__ xq,
                                                    float* __restrict__ rsc) {
    __shared__ float red[256];
    const int row = blockIdx.x, t = threadIdx.x;
    const float* xr = x + (size_t)row * DIN;
    float xv[4];
    float ss = 0.f;
    #pragma unroll
    for (int it = 0; it < 4; ++it) {
        int i = t + it * 256;
        float v = (i < DIN) ? xr[i] : 0.f;
        xv[it] = v; ss += v * v;
    }
    red[t] = ss; __syncthreads();
    for (int s = 128; s > 0; s >>= 1) {
        if (t < s) red[t] += red[t + s];
        __syncthreads();
    }
    float rn = rsqrtf(red[0] / (float)DIN + RMS_EPS);
    __syncthreads();
    float am = 0.f;
    #pragma unroll
    for (int it = 0; it < 4; ++it) am = fmaxf(am, fabsf(xv[it] * rn));
    red[t] = am; __syncthreads();
    for (int s = 128; s > 0; s >>= 1) {
        if (t < s) red[t] = fmaxf(red[t], red[t + s]);
        __syncthreads();
    }
    float clipv = fmaxf(red[0], Q_EPS);
    float sc = 127.f / clipv;
    #pragma unroll
    for (int it = 0; it < 4; ++it) {
        int i = t + it * 256;
        if (i < KP1) {
            float q = (i < DIN)
                ? fminf(fmaxf(rintf(xv[it] * rn * sc), -128.f), 127.f) : 0.f;
            xq[(size_t)row * KP1 + i] = (int8_t)q;
        }
    }
    if (t == 0) rsc[row] = clipv / 127.f;
}

__global__ __launch_bounds__(1024) void k_gemm1(const int8_t* __restrict__ xq,
                                                const int8_t* __restrict__ w1q,
                                                const float* __restrict__ rsc,
                                                const float* __restrict__ b1,
                                                const float* __restrict__ wsum,
                                                int8_t* __restrict__ hq,
                                                float* __restrict__ hscale) {
    __shared__ unsigned int As[256];
    __shared__ float    s_ss[16];
    __shared__ unsigned s_am[16];
    const int tid  = threadIdx.x;
    const int wave = tid >> 5, lane = tid & 31;
    const int half = lane >> 4, l15 = lane & 15;
    const int rows0 = blockIdx.x * 16;
    const int cols0 = wave * 128;

    v8i acc[8];
    #pragma unroll
    for (int nt = 0; nt < 8; ++nt)
        #pragma unroll
        for (int j = 0; j < 8; ++j) acc[nt][j] = 0;
    if (tid < 16) { s_ss[tid] = 0.f; s_am[tid] = 0u; }

    for (int ks = 0; ks < 13; ++ks) {
        const int k0 = ks * 64;
        __syncthreads();
        if (tid < 256) {
            int r = tid >> 4, c = tid & 15;
            As[tid] = *(const unsigned int*)(xq + (size_t)(rows0 + r) * KP1 + k0 + c * 4);
        }
        __syncthreads();
        v8i a;
        {
            const int base = l15 * 16 + half * 4;
            #pragma unroll
            for (int p = 0; p < 4; ++p) {
                uint2 t = *(const uint2*)&As[base + (p >> 1) * 8 + (p & 1) * 2];
                a[2 * p] = (int)t.x; a[2 * p + 1] = (int)t.y;
            }
        }
        #pragma unroll
        for (int nt = 0; nt < 8; ++nt) {
            const size_t boff = (size_t)(cols0 + nt * 16 + l15) * KP1 + k0 + half * 16;
            uint4 t0 = *(const uint4*)(w1q + boff);
            uint4 t1 = *(const uint4*)(w1q + boff + 32);
            v8i b;
            b[0] = (int)t0.x; b[1] = (int)t0.y; b[2] = (int)t0.z; b[3] = (int)t0.w;
            b[4] = (int)t1.x; b[5] = (int)t1.y; b[6] = (int)t1.z; b[7] = (int)t1.w;
            acc[nt] = __builtin_amdgcn_wmma_i32_16x16x64_iu8(
                true, a, true, b, acc[nt], false, false);
            if (ks < 12) __builtin_prefetch(w1q + boff + 64, 0, 1);
        }
    }
    __syncthreads();

    const float m1 = fmaxf(wsum[0] / (float)(DH * DIN), Q_EPS);
    float rsv[8], bc[8];
    #pragma unroll
    for (int j = 0; j < 8; ++j) rsv[j] = rsc[rows0 + j + 8 * half] * m1;
    #pragma unroll
    for (int nt = 0; nt < 8; ++nt) bc[nt] = b1[cols0 + nt * 16 + l15];

    float ssl[8], aml[8];
    #pragma unroll
    for (int j = 0; j < 8; ++j) { ssl[j] = 0.f; aml[j] = 0.f; }
    #pragma unroll
    for (int nt = 0; nt < 8; ++nt)
        #pragma unroll
        for (int j = 0; j < 8; ++j) {
            float v = fmaxf((float)acc[nt][j] * rsv[j] + bc[nt], 0.f);
            ssl[j] += v * v;
            aml[j] = fmaxf(aml[j], v);
        }
    #pragma unroll
    for (int j = 0; j < 8; ++j) {
        int rl = j + 8 * half;
        atomicAdd(&s_ss[rl], ssl[j]);
        atomicMax(&s_am[rl], __float_as_uint(aml[j]));
    }
    __syncthreads();

    #pragma unroll
    for (int j = 0; j < 8; ++j) {
        int rl = j + 8 * half;
        float rn    = rsqrtf(s_ss[rl] / (float)DH + RMS_EPS);
        float clipv = fmaxf(__uint_as_float(s_am[rl]) * rn, Q_EPS);
        float sc    = 127.f / clipv;
        size_t rowoff = (size_t)(rows0 + rl) * KP2;
        #pragma unroll
        for (int nt = 0; nt < 8; ++nt) {
            float v = fmaxf((float)acc[nt][j] * rsv[j] + bc[nt], 0.f);
            float q = fminf(fmaxf(rintf(v * rn * sc), -128.f), 127.f);
            hq[rowoff + cols0 + nt * 16 + l15] = (int8_t)q;
        }
    }
    if (tid < 16) {
        float rn    = rsqrtf(s_ss[tid] / (float)DH + RMS_EPS);
        float clipv = fmaxf(__uint_as_float(s_am[tid]) * rn, Q_EPS);
        hscale[rows0 + tid] = clipv / 127.f;
    }
}

__global__ __launch_bounds__(256) void k_gemm2(const int8_t* __restrict__ hq,
                                               const int8_t* __restrict__ w2q,
                                               const float* __restrict__ hscale,
                                               const float* __restrict__ b2,
                                               const float* __restrict__ wsum2,
                                               float* __restrict__ out) {
    const int tid  = threadIdx.x;
    const int wave = tid >> 5, lane = tid & 31;
    const int half = lane >> 4, l15 = lane & 15;
    const int rows0 = (blockIdx.x * 8 + wave) * 16;

    v8i acc;
    #pragma unroll
    for (int j = 0; j < 8; ++j) acc[j] = 0;

    const size_t arow = (size_t)(rows0 + l15) * KP2;
    for (int ks = 0; ks < 64; ++ks) {
        const int k0 = ks * 64;
        v8i a;
        #pragma unroll
        for (int p = 0; p < 4; ++p) {
            uint2 t = *(const uint2*)(hq + arow + k0 + (p >> 1) * 32 + half * 16 + (p & 1) * 8);
            a[2 * p] = (int)t.x; a[2 * p + 1] = (int)t.y;
        }
        const size_t boff = (size_t)l15 * KP2 + k0 + half * 16;
        uint4 t0 = *(const uint4*)(w2q + boff);
        uint4 t1 = *(const uint4*)(w2q + boff + 32);
        v8i b;
        b[0] = (int)t0.x; b[1] = (int)t0.y; b[2] = (int)t0.z; b[3] = (int)t0.w;
        b[4] = (int)t1.x; b[5] = (int)t1.y; b[6] = (int)t1.z; b[7] = (int)t1.w;
        acc = __builtin_amdgcn_wmma_i32_16x16x64_iu8(true, a, true, b, acc, false, false);
    }
    const float m2 = fmaxf(wsum2[0] / (float)(DOUT * DH), Q_EPS);
    const int n = l15;
    if (n < DOUT) {
        #pragma unroll
        for (int j = 0; j < 8; ++j) {
            int r = rows0 + j + 8 * half;
            float v = (float)acc[j] * hscale[r] * m2 + b2[n];
            out[(size_t)r * DOUT + n] = v;
        }
    }
}


extern "C" void kernel_launch(void* const* d_in, const int* in_sizes, int n_in,
                              void* d_out, int out_size, void* d_ws, size_t ws_size,
                              hipStream_t stream) {
    const float* x  = (const float*)d_in[0];
    const float* w1 = (const float*)d_in[1];
    const float* b1 = (const float*)d_in[2];
    const float* w2 = (const float*)d_in[3];
    const float* b2 = (const float*)d_in[4];
    float* out = (float*)d_out;

    char* ws = (char*)d_ws;
    size_t off = 0;
    float*  sums = (float*)(ws + off); off += 256;
    float*  rscx = (float*)(ws + off); off += (size_t)BATCH * 4;
    float*  hsc  = (float*)(ws + off); off += (size_t)BATCH * 4;
    int8_t* xq   = (int8_t*)(ws + off); off += (size_t)BATCH * KP1;
    int8_t* w1q  = (int8_t*)(ws + off); off += (size_t)DH * KP1;
    int8_t* w2q  = (int8_t*)(ws + off); off += (size_t)NP2 * KP2;
    int8_t* hq   = (int8_t*)(ws + off); off += (size_t)BATCH * KP2;
    (void)ws_size; (void)in_sizes; (void)n_in; (void)out_size;

    k_zero2<<<1, 32, 0, stream>>>(sums);
    k_absum<<<2048, 256, 0, stream>>>(w1, DH * DIN, &sums[0]);
    k_absum<<<160, 256, 0, stream>>>(w2, DOUT * DH, &sums[1]);
    k_wq1<<<(DH * KP1 + 255) / 256, 256, 0, stream>>>(w1, &sums[0], w1q);
    k_wq2<<<(NP2 * KP2 + 255) / 256, 256, 0, stream>>>(w2, &sums[1], w2q);
    k_rowquant_x<<<BATCH, 256, 0, stream>>>(x, xq, rscx);
    k_gemm1<<<BATCH / 16, 1024, 0, stream>>>(xq, w1q, rscx, b1, &sums[0], hq, hsc);
    k_gemm2<<<BATCH / (16 * 8), 256, 0, stream>>>(hq, w2q, hsc, b2, &sums[1], out);
}
